// Cluster_51642686767978
// MI455X (gfx1250) — matrix operand order repaired and checked by reading, not compiled, not run on hardware
//
#include <hip/hip_runtime.h>
#include <hip/hip_bf16.h>
#include <math.h>

typedef _Float16 half_t;
typedef __attribute__((ext_vector_type(16))) _Float16 v16h;
typedef __attribute__((ext_vector_type(8)))  _Float16 v8h;
typedef __attribute__((ext_vector_type(8)))  float    v8f;

#define NB   4          // batch
#define TT   1024       // time
#define FF   252        // freq bins
#define FOQ  84         // freq bins after stride-3 convs
#define NPOS (FOQ*TT)   // 86016

__constant__ int c_shifts[8] = {-36, 0, 36, 57, 72, 84, 93, 101};
__constant__ int c_sel[12]   = {0, 2, 3, 4, 5, 6, 7, 8, 9, 10, 12, 14};

// ---------------------------------------------------------------- init / amax
__global__ void k_init(float* ws_f) {
    for (int i = threadIdx.x; i < 4 + NB * 18 * 18; i += blockDim.x) ws_f[i] = 0.f;
}

__global__ __launch_bounds__(256) void k_amax(const float* __restrict__ x, float* amax) {
    __shared__ float red[256];
    const int b = blockIdx.y;
    const int i = blockIdx.x * 256 + threadIdx.x;   // over FF*TT = 258048
    float a = 0.f;
    if (i < FF * TT) {
        float x0 = x[(b * 2 + 0) * FF * TT + i];
        float x1 = x[(b * 2 + 1) * FF * TT + i];
        a = sqrtf(x0 * x0 + x1 * x1 + 1e-8f);
    }
    red[threadIdx.x] = a; __syncthreads();
    for (int s = 128; s > 0; s >>= 1) {
        if (threadIdx.x < s) red[threadIdx.x] = fmaxf(red[threadIdx.x], red[threadIdx.x + s]);
        __syncthreads();
    }
    if (threadIdx.x == 0)
        atomicMax((unsigned int*)&amax[b], __float_as_uint(red[0]));
}

// ---------------------------------------------------------------- HCQT stack
__global__ __launch_bounds__(256) void k_stack(const float* __restrict__ x,
                                               const float* __restrict__ amax,
                                               half_t* __restrict__ out) {
    int idx = blockIdx.x * 256 + threadIdx.x;
    if (idx >= NB * 16 * FF * TT) return;
    int t = idx & (TT - 1);
    int f = (idx / TT) % FF;
    int c = (idx / (TT * FF)) % 16;
    int b = idx / (TT * FF * 16);
    int h = c >> 1, p = c & 1;
    int fs = f + c_shifts[h];
    float v = 0.f;
    if (fs >= 0 && fs < FF) {
        float x0 = x[((b * 2 + 0) * FF + fs) * TT + t];
        float x1 = x[((b * 2 + 1) * FF + fs) * TT + t];
        if (p) v = atan2f(x1, x0);
        else {
            float amp = sqrtf(x0 * x0 + x1 * x1 + 1e-8f);
            v = amp / (amax[b] + 1e-8f);
        }
    }
    out[idx] = (half_t)v;
}

// ---------------------------------------------------------------- weight pre-pack into A-fragment order
// Wp[((mt*KC + kc)*32 + lane)*16 + j]
template <int CIN, int COUT, int KH, int KW>
__global__ __launch_bounds__(256) void k_wpack(const float* __restrict__ W,
                                               half_t* __restrict__ Wp) {
    constexpr int KTOT = CIN * KH * KW;
    constexpr int KC = (KTOT + 31) / 32;
    int idx = blockIdx.x * 256 + threadIdx.x;
    if (idx >= 2 * KC * 512) return;
    int j = idx & 15;
    int lane = (idx >> 4) & 31;
    int rest = idx >> 9;
    int kc = rest % KC;
    int mt = rest / KC;
    int m = mt * 16 + (lane & 15);
    int kb = (lane >> 4) * 16;
    int k = kc * 32 + kb + j;
    float v = 0.f;
    if (m < COUT && k < KTOT) {
        int kw = k % KW, kh = (k / KW) % KH, ci = k / (KW * KH);
        v = W[((m * CIN + ci) * KH + kh) * KW + kw];
    }
    Wp[idx] = (half_t)v;
}

// ---------------------------------------------------------------- WMMA implicit-GEMM conv
// 4 waves/block: 2 m-tiles x 2 t-subtiles sharing one LDS input tile.
// ACT: 0 none, 1 leaky_relu(0.01), 2 relu, 3 silu
template <int CIN, int COUT, int KH, int KW, int PH, int PW, int SH, int DH,
          int ACT, bool OUT16, bool ACCUM, bool HASBN>
__global__ __launch_bounds__(128)
void conv_wmma(const half_t* __restrict__ in, const half_t* __restrict__ Wp,
               const float* __restrict__ bias, const float* __restrict__ g,
               const float* __restrict__ bt, half_t* __restrict__ out16,
               float* __restrict__ out32, int Fin, int Fout, int CO_TOT, int CBASE) {
    constexpr int KTOT = CIN * KH * KW;
    constexpr int KC = (KTOT + 31) / 32;
    constexpr int TLOC = 36;                       // >= 31 + KW
    __shared__ half_t sIn[CIN * KH * TLOC];
    __shared__ unsigned short sOff[KC * 32];

    const int tid = threadIdx.x;
    const int lane = tid & 31;
    const int wv = tid >> 5;
    const int wt = wv & 1;                         // time subtile
    const int mt = wv >> 1;                        // output-channel tile
    const int tblk = blockIdx.x * 32;
    const int fo = blockIdx.y;
    const int b = blockIdx.z;

    // stage input footprint (zeros at pad)
    for (int i = tid; i < CIN * KH * TLOC; i += 128) {
        int tl = i % TLOC;
        int row = i / TLOC;
        int kh = row % KH;
        int ci = row / KH;
        int fi = fo * SH - PH + kh * DH;
        int gt = tblk - PW + tl;
        half_t v = (half_t)0.f;
        if (tl < 31 + KW && fi >= 0 && fi < Fin && gt >= 0 && gt < TT) {
            const half_t* p = &in[((b * CIN + ci) * Fin + fi) * TT + gt];
            v = *p;
            if (tl == 0 && fi + SH < Fin)
                __builtin_prefetch(p + SH * TT, 0, 1);   // next output row -> global_prefetch_b8
        }
        sIn[i] = v;
    }
    // k -> LDS offset LUT (sentinel for K padding)
    for (int k = tid; k < KC * 32; k += 128) {
        unsigned short o = 0xFFFFu;
        if (k < KTOT) {
            int kw = k % KW, kh = (k / KW) % KH, ci = k / (KW * KH);
            o = (unsigned short)((ci * KH + kh) * TLOC + kw);
        }
        sOff[k] = o;
    }
    __syncthreads();

    const int nB = lane & 15;
    const int koffB = (lane >> 4) * 16;
    const int tl_base = wt * 16 + nB;

    v8f acc = {};
    const v16h* Wpv = (const v16h*)Wp;
    for (int kc = 0; kc < KC; ++kc) {
        v16h a = Wpv[(mt * KC + kc) * 32 + lane];  // pre-packed A fragment
        v16h bf;
#pragma unroll
        for (int j = 0; j < 16; ++j) {
            unsigned short o = sOff[kc * 32 + koffB + j];
            bf[j] = (o == 0xFFFFu) ? (half_t)0.f : sIn[o + tl_base];
        }
        acc = __builtin_amdgcn_wmma_f32_16x16x32_f16(false, a, false, bf,
                                                     (short)0, acc, false, false);
    }

    const int tB = tblk + wt * 16 + nB;
#pragma unroll
    for (int r = 0; r < 8; ++r) {
        int m = mt * 16 + r + ((lane >> 4) << 3);
        if (m >= COUT) continue;
        float v = acc[r] + bias[m];
        if (HASBN) v = v * g[m] + bt[m];
        if (ACT == 1) v = v > 0.f ? v : 0.01f * v;
        else if (ACT == 2) v = fmaxf(v, 0.f);
        else if (ACT == 3) v = v / (1.f + expf(-v));
        int oidx = ((b * CO_TOT + CBASE + m) * Fout + fo) * TT + tB;
        if (OUT16) out16[oidx] = (half_t)v;
        else { if (ACCUM) out32[oidx] += v; else out32[oidx] = v; }
    }
}

// ---------------------------------------------------------------- SEL channel copy
__global__ __launch_bounds__(256) void k_selcopy(const half_t* __restrict__ st,
                                                 half_t* __restrict__ ec) {
    int idx = blockIdx.x * 256 + threadIdx.x;
    if (idx >= NB * 12 * FF * TT) return;
    int ft = idx % (FF * TT);
    int j = (idx / (FF * TT)) % 12;
    int b = idx / (FF * TT * 12);
    ec[(b * 32 + j) * FF * TT + ft] = st[(b * 16 + c_sel[j]) * FF * TT + ft];
}

// ---------------------------------------------------------------- mask conv (25->1, 5x5, stride (3,1))
__global__ __launch_bounds__(256) void k_mask(const half_t* __restrict__ neck,
                                              const float* __restrict__ Wm,
                                              const float* __restrict__ bm,
                                              float* __restrict__ mask) {
    int idx = blockIdx.x * 256 + threadIdx.x;
    if (idx >= NB * NPOS) return;
    int t = idx & (TT - 1);
    int fo = (idx / TT) % FOQ;
    int b = idx / NPOS;
    float acc = bm[0];
    for (int kh = 0; kh < 5; ++kh) {
        int fi = fo * 3 - 1 + kh;
        if (fi < 0 || fi >= FF) continue;
        for (int kw = 0; kw < 5; ++kw) {
            int ti = t - 2 + kw;
            if (ti < 0 || ti >= TT) continue;
            float ss = 0.f;
            for (int c = 0; c < 24; ++c) {
                float v = (float)neck[((b * 24 + c) * FF + fi) * TT + ti];
                acc += Wm[c * 25 + kh * 5 + kw] * v;
                ss += v * v;
            }
            acc += Wm[24 * 25 + kh * 5 + kw] * ss;
        }
    }
    mask[idx] = 1.f / (1.f + expf(-acc));
}

// ---------------------------------------------------------------- neck += pe
__global__ __launch_bounds__(256) void k_addpe(half_t* __restrict__ neck,
                                               const float* __restrict__ pe) {
    int idx = blockIdx.x * 256 + threadIdx.x;
    if (idx >= NB * 24 * FF * TT) return;
    int f = (idx / TT) % FF;
    int c = (idx / (TT * FF)) % 24;
    neck[idx] = (half_t)((float)neck[idx] + pe[c * FF + f]);
}

// ---------------------------------------------------------------- masknorm (fp16 buffers, K/V)
__global__ __launch_bounds__(256) void k_masknorm_h(half_t* __restrict__ buf,
                                                    const float* __restrict__ mask) {
    int i = blockIdx.x * 256 + threadIdx.x;
    if (i >= NB * NPOS) return;
    int b = i / NPOS, pos = i % NPOS;
    float ss = 0.f;
    for (int c = 0; c < 18; ++c) {
        float v = (float)buf[(b * 18 + c) * NPOS + pos];
        ss += v * v;
    }
    float sc = mask[i] / sqrtf(ss + 1e-8f);
    for (int c = 0; c < 18; ++c) {
        int idx = (b * 18 + c) * NPOS + pos;
        buf[idx] = (half_t)((float)buf[idx] * sc);
    }
}

// ---------------------------------------------------------------- ctx = K V^T via WMMA
// GEMM: M=N=18 (padded to 32), K = NPOS. Fragments are direct vector loads
// because the reduction dim is contiguous in memory for both operands.
__global__ __launch_bounds__(128) void k_ctx_wmma(const half_t* __restrict__ K,
                                                  const half_t* __restrict__ V,
                                                  float* __restrict__ ctx) {
    const int tid = threadIdx.x, lane = tid & 31, wv = tid >> 5;
    const int mt = wv >> 1, nt = wv & 1;
    const int b = blockIdx.y;
    const int k0 = blockIdx.x * 1024;              // NPOS / 1024 = 84 slabs
    int mrow = mt * 16 + (lane & 15); if (mrow >= 18) mrow = 0;   // clamp: garbage rows discarded
    int ncol = nt * 16 + (lane & 15); if (ncol >= 18) ncol = 0;
    const int kbA = (lane >> 4) * 16;
    const int koffB = (lane >> 4) * 16;
    const half_t* Arow = K + (b * 18 + mrow) * NPOS;
    const half_t* Bcol = V + (b * 18 + ncol) * NPOS;

    v8f acc = {};
    for (int kk = 0; kk < 1024; kk += 32) {
        const int kbase = k0 + kk;
        v8h alo = *(const v8h*)(Arow + kbase + kbA);
        v8h ahi = *(const v8h*)(Arow + kbase + kbA + 8);
        v16h a, bf;
#pragma unroll
        for (int j = 0; j < 8; ++j) { a[j] = alo[j]; a[j + 8] = ahi[j]; }
        bf = *(const v16h*)(Bcol + kbase + koffB);
        acc = __builtin_amdgcn_wmma_f32_16x16x32_f16(false, a, false, bf,
                                                     (short)0, acc, false, false);
    }
#pragma unroll
    for (int r = 0; r < 8; ++r) {
        int m = mt * 16 + r + ((lane >> 4) << 3);
        int n = nt * 16 + (lane & 15);
        if (m < 18 && n < 18) atomicAdd(&ctx[b * 324 + m * 18 + n], acc[r]);
    }
}

// ---------------------------------------------------------------- la = ctx^T Q (tiny; fp32 for precision)
__global__ __launch_bounds__(256) void k_la(const float* __restrict__ ctx,
                                            const half_t* __restrict__ Q,
                                            half_t* __restrict__ la) {
    __shared__ float sC[324];
    const int i = blockIdx.x * 256 + threadIdx.x;   // NPOS % 256 == 0, blocks never straddle b
    const int b = i / NPOS, pos = i % NPOS;
    for (int j = threadIdx.x; j < 324; j += 256) sC[j] = ctx[b * 324 + j];
    __syncthreads();
    float q[18];
    for (int d = 0; d < 18; ++d) q[d] = (float)Q[(b * 18 + d) * NPOS + pos];
    for (int c = 0; c < 18; ++c) {
        float s = 0.f;
        for (int d = 0; d < 18; ++d) s += sC[d * 18 + c] * q[d];
        la[(b * 18 + c) * NPOS + pos] = (half_t)s;
    }
}

// ---------------------------------------------------------------- final channel norm on f32 emb
__global__ __launch_bounds__(256) void k_norm_f(float* __restrict__ emb) {
    int i = blockIdx.x * 256 + threadIdx.x;
    if (i >= NB * NPOS) return;
    int b = i / NPOS, pos = i % NPOS;
    float ss = 0.f;
    for (int c = 0; c < 18; ++c) {
        float v = emb[(b * 18 + c) * NPOS + pos];
        ss += v * v;
    }
    float sc = 1.f / sqrtf(ss + 1e-8f);
    for (int c = 0; c < 18; ++c) emb[(b * 18 + c) * NPOS + pos] *= sc;
}

// ================================================================ launch
extern "C" void kernel_launch(void* const* d_in, const int* in_sizes, int n_in,
                              void* d_out, int out_size, void* d_ws, size_t ws_size,
                              hipStream_t stream) {
    (void)in_sizes; (void)n_in; (void)out_size; (void)ws_size;
    const float* x   = (const float*)d_in[0];
    const float* W1  = (const float*)d_in[1];  const float* b1  = (const float*)d_in[2];
    const float* g1  = (const float*)d_in[3];  const float* bt1 = (const float*)d_in[4];
    const float* W2  = (const float*)d_in[5];  const float* b2  = (const float*)d_in[6];
    const float* g2  = (const float*)d_in[7];  const float* bt2 = (const float*)d_in[8];
    const float* Wn  = (const float*)d_in[9];  const float* bnb = (const float*)d_in[10];
    const float* gn  = (const float*)d_in[11]; const float* btn = (const float*)d_in[12];
    const float* Wm  = (const float*)d_in[13]; const float* bm  = (const float*)d_in[14];
    const float* pe  = (const float*)d_in[15];
    const float* Wq  = (const float*)d_in[16]; const float* bq  = (const float*)d_in[17];
    const float* Wk  = (const float*)d_in[18]; const float* bk  = (const float*)d_in[19];
    const float* Wv  = (const float*)d_in[20]; const float* bv  = (const float*)d_in[21];
    const float* Wa  = (const float*)d_in[22]; const float* ba  = (const float*)d_in[23];
    const float* We  = (const float*)d_in[24]; const float* be  = (const float*)d_in[25];

    // workspace layout (bytes)
    char* ws = (char*)d_ws;
    float* ws_f = (float*)d_ws;
    float* amax = ws_f;            // 4 floats
    float* ctx  = ws_f + 4;        // 4*324 floats
    constexpr size_t HDR  = 8192;
    constexpr size_t S16  = (size_t)NB * 16 * FF * TT * 2;  // 33 MB
    constexpr size_t S24  = (size_t)NB * 24 * FF * TT * 2;  // 49.5 MB
    constexpr size_t SQ   = (size_t)NB * 18 * NPOS * 2;     // 12.4 MB
    half_t* stacked = (half_t*)(ws + HDR);
    half_t* h1      = (half_t*)(ws + HDR + S16);
    half_t* ec      = (half_t*)(ws + HDR + S16 + S24);
    half_t* neck    = (half_t*)(ws + HDR);                  // aliases stacked+h1 (dead by then)
    half_t* Qb      = (half_t*)(ws + HDR + S16 + S24);      // aliases ec (dead by then)
    half_t* Kb      = Qb + (size_t)NB * 18 * NPOS;
    half_t* Vb      = Kb + (size_t)NB * 18 * NPOS;
    half_t* la      = Vb + (size_t)NB * 18 * NPOS;
    // packed weights (A-fragment order), after la; all offsets 32B aligned
    half_t* wp      = la + (size_t)NB * 18 * NPOS;
    half_t* wp1 = wp;                 // 2*13*512
    half_t* wp2 = wp1 + 2 * 13 * 512; // 2*57*512
    half_t* wpn = wp2 + 2 * 57 * 512; // 2*25*512
    half_t* wpq = wpn + 2 * 25 * 512; // 2*7*512
    half_t* wpk = wpq + 2 * 7 * 512;
    half_t* wpv = wpk + 2 * 7 * 512;
    half_t* wpa = wpv + 2 * 7 * 512;  // 2*2*512
    half_t* wpe = wpa + 2 * 2 * 512;  // 2*12*512

    float* emb  = (float*)d_out;                       // (4,18,84,1024)
    float* mask = (float*)d_out + NB * 18 * NPOS;      // (4,84,1024)

    // 0) zero amax + ctx; pack all weights into fragment order
    k_init<<<1, 256, 0, stream>>>(ws_f);
    k_wpack<16, 24, 5, 5><<<(2 * 13 * 512 + 255) / 256, 256, 0, stream>>>(W1, wp1);
    k_wpack<24, 20, 25, 3><<<(2 * 57 * 512 + 255) / 256, 256, 0, stream>>>(W2, wp2);
    k_wpack<32, 24, 5, 5><<<(2 * 25 * 512 + 255) / 256, 256, 0, stream>>>(Wn, wpn);
    k_wpack<24, 18, 3, 3><<<(2 * 7 * 512 + 255) / 256, 256, 0, stream>>>(Wq, wpq);
    k_wpack<24, 18, 3, 3><<<(2 * 7 * 512 + 255) / 256, 256, 0, stream>>>(Wk, wpk);
    k_wpack<24, 18, 3, 3><<<(2 * 7 * 512 + 255) / 256, 256, 0, stream>>>(Wv, wpv);
    k_wpack<18, 18, 1, 3><<<(2 * 2 * 512 + 255) / 256, 256, 0, stream>>>(Wa, wpa);
    k_wpack<24, 18, 5, 3><<<(2 * 12 * 512 + 255) / 256, 256, 0, stream>>>(We, wpe);
    // 1) per-batch amplitude max
    k_amax<<<dim3((FF * TT + 255) / 256, NB), 256, 0, stream>>>(x, amax);
    // 2) HCQT stack -> (4,16,252,1024) f16
    k_stack<<<(NB * 16 * FF * TT) / 256, 256, 0, stream>>>(x, amax, stacked);
    // 3) conv1: 16->24, 5x5 pad2, BN + leaky_relu
    conv_wmma<16, 24, 5, 5, 2, 2, 1, 1, 1, true, false, true>
        <<<dim3(TT / 32, FF, NB), 128, 0, stream>>>(stacked, wp1, b1, g1, bt1, h1, nullptr, FF, FF, 24, 0);
    // 4) conv2: 24->20, 25x3 dil(3,1) pad(36,1), BN + relu -> ec[12:32]
    conv_wmma<24, 20, 25, 3, 36, 1, 1, 3, 2, true, false, true>
        <<<dim3(TT / 32, FF, NB), 128, 0, stream>>>(h1, wp2, b2, g2, bt2, ec, nullptr, FF, FF, 32, 12);
    // 5) stacked[SEL] -> ec[0:12]
    k_selcopy<<<(NB * 12 * FF * TT) / 256, 256, 0, stream>>>(stacked, ec);
    // 6) neck conv: 32->24, 5x5 pad2, BN + silu
    conv_wmma<32, 24, 5, 5, 2, 2, 1, 1, 3, true, false, true>
        <<<dim3(TT / 32, FF, NB), 128, 0, stream>>>(ec, wpn, bnb, gn, btn, neck, nullptr, FF, FF, 24, 0);
    // 7) mask conv (scalar; 0.43 GFLOP) -> d_out mask region
    k_mask<<<(NB * NPOS) / 256, 256, 0, stream>>>(neck, Wm, bm, mask);
    // 8) neck += pe
    k_addpe<<<(NB * 24 * FF * TT) / 256, 256, 0, stream>>>(neck, pe);
    // 9) Q/K/V convs: 24->18, 3x3, stride(3,1), pad(0,1)
    conv_wmma<24, 18, 3, 3, 0, 1, 3, 1, 0, true, false, false>
        <<<dim3(TT / 32, FOQ, NB), 128, 0, stream>>>(neck, wpq, bq, nullptr, nullptr, Qb, nullptr, FF, FOQ, 18, 0);
    conv_wmma<24, 18, 3, 3, 0, 1, 3, 1, 0, true, false, false>
        <<<dim3(TT / 32, FOQ, NB), 128, 0, stream>>>(neck, wpk, bk, nullptr, nullptr, Kb, nullptr, FF, FOQ, 18, 0);
    conv_wmma<24, 18, 3, 3, 0, 1, 3, 1, 0, true, false, false>
        <<<dim3(TT / 32, FOQ, NB), 128, 0, stream>>>(neck, wpv, bv, nullptr, nullptr, Vb, nullptr, FF, FOQ, 18, 0);
    // 10) masknorm K, V
    k_masknorm_h<<<(NB * NPOS) / 256, 256, 0, stream>>>(Kb, mask);
    k_masknorm_h<<<(NB * NPOS) / 256, 256, 0, stream>>>(Vb, mask);
    // 11) ctx = K V^T via WMMA (84 slabs x 4 batches, fp32 atomic accumulate)
    k_ctx_wmma<<<dim3(NPOS / 1024, NB), 128, 0, stream>>>(Kb, Vb, ctx);
    // 12) la = ctx^T Q
    k_la<<<(NB * NPOS) / 256, 256, 0, stream>>>(ctx, Qb, la);
    // 13) emb = conv(la, Wa, 1x3 pad(0,1)) ...
    conv_wmma<18, 18, 1, 3, 0, 1, 1, 1, 0, false, false, false>
        <<<dim3(TT / 32, FOQ, NB), 128, 0, stream>>>(la, wpa, ba, nullptr, nullptr, nullptr, emb, FOQ, FOQ, 18, 0);
    //     ... + conv(neck, We, 5x3 stride(3,1) pad(1,1))  (accumulate)
    conv_wmma<24, 18, 5, 3, 1, 1, 3, 1, 0, false, true, false>
        <<<dim3(TT / 32, FOQ, NB), 128, 0, stream>>>(neck, wpe, be, nullptr, nullptr, nullptr, emb, FF, FOQ, 18, 0);
    // 14) final unit-norm over channels
    k_norm_f<<<(NB * NPOS) / 256, 256, 0, stream>>>(emb);
}
